// GatModule_48387101557057
// MI455X (gfx1250) — matrix operand order repaired and checked by reading, not compiled, not run on hardware
//
#include <hip/hip_runtime.h>

typedef __attribute__((ext_vector_type(16))) _Float16 v16h;
typedef __attribute__((ext_vector_type(4)))  _Float16 v4h;
typedef __attribute__((ext_vector_type(8)))  float    v8f;
typedef __attribute__((ext_vector_type(4)))  int      v4i;

#define IN_F   256   // IN_FEATS
#define HF     256   // NUM_HEADS * OUT_FEATS
#define NHEAD  8
#define OUTF   32
#define NEG_SLOPE 0.2f

#if __has_builtin(__builtin_amdgcn_global_load_async_to_lds_b128) && \
    __has_builtin(__builtin_amdgcn_s_wait_asynccnt)
#define USE_ASYNC_LDS 1
typedef __attribute__((address_space(1))) v4i* gas_v4i;   // global
typedef __attribute__((address_space(3))) v4i* lds_v4i;   // LDS
#else
#define USE_ASYNC_LDS 0
#endif

// ---------------------------------------------------------------------------
// helpers
// ---------------------------------------------------------------------------
__device__ __forceinline__ float leaky(float x) {
    return x > 0.0f ? x : NEG_SLOPE * x;
}
// order-preserving float -> uint encoding for integer atomicMax
__device__ __forceinline__ unsigned enc_f32(float x) {
    unsigned b = __float_as_uint(x);
    return (b & 0x80000000u) ? ~b : (b | 0x80000000u);
}
__device__ __forceinline__ float dec_f32(unsigned k) {
    unsigned b = (k & 0x80000000u) ? (k & 0x7FFFFFFFu) : ~k;
    return __uint_as_float(b);
}
__device__ __forceinline__ void atomic_add_f32(float* p, float v) {
    __hip_atomic_fetch_add(p, v, __ATOMIC_RELAXED, __HIP_MEMORY_SCOPE_AGENT);
}

// ---------------------------------------------------------------------------
// K0: zero output + softmax accumulators (harness poisons buffers, and we
// accumulate with atomics, so re-init every call)
// ---------------------------------------------------------------------------
__global__ void gat_init(float* __restrict__ out, unsigned* __restrict__ m_enc,
                         float* __restrict__ denom, size_t n_out, size_t n_nh) {
    size_t i = (size_t)blockIdx.x * blockDim.x + threadIdx.x;
    size_t stride = (size_t)gridDim.x * blockDim.x;
    for (size_t j = i; j < n_out; j += stride) out[j] = 0.0f;
    for (size_t j = i; j < n_nh; j += stride) { m_enc[j] = 0u; denom[j] = 0.0f; }
}

// ---------------------------------------------------------------------------
// K0.5: W (256x256 fp32) -> fp16 copy; tiny (128 KB), amortized over 3125
// GEMM blocks and enables raw async byte-copy staging of W slabs into LDS.
// ---------------------------------------------------------------------------
__global__ __launch_bounds__(256) void gat_convert_w(const float* __restrict__ W,
                                                     _Float16* __restrict__ Wh) {
    int i = blockIdx.x * blockDim.x + threadIdx.x;   // 0..65535
    Wh[i] = (_Float16)W[i];
}

// ---------------------------------------------------------------------------
// K1: ft = feat @ W.T   (M x 256, K = 256), fp16 WMMA, fp32 accum
// block = 128 threads = 4 waves; block tile = 16 (M) x 256 (N)
// wave w covers N columns [w*64, w*64+64) as four 16x16 WMMA tiles
// W slab staged into LDS via GLOBAL_LOAD_ASYNC_TO_LDS_B128 when available.
// ---------------------------------------------------------------------------
__global__ __launch_bounds__(128) void gat_gemm(const float* __restrict__ feat,
                                                const _Float16* __restrict__ Wh,
                                                float* __restrict__ ft, int M) {
    __shared__ _Float16 Bs[HF * 32];          // W K-slab fp16: [n][k], 16 KB

    const int m0   = blockIdx.x * 16;
    const int tid  = threadIdx.x;
    const int wave = tid >> 5;                // 0..3
    const int lane = tid & 31;
    const int n0   = wave * 64;

    v8f acc0 = {}, acc1 = {}, acc2 = {}, acc3 = {};

    // A-fragment addressing, contiguous k order used here for A and B (16x32, M x K):
    // lanes 0-15: M=lane, K = {0..15}
    // lanes 16-31: M=lane-16, K = {16..31}
    int mrow = m0 + (lane & 15);
    if (mrow >= M) mrow = M - 1;              // clamp loads; stores predicated
    const int akb = (lane >> 4) * 16;         // 0 or 16

    for (int kk = 0; kk < IN_F; kk += 32) {
        __syncthreads();                      // Bs reuse from previous iter done
        // --- stage Wh[n][kk..kk+32) into LDS: 1024 x 16B granules ---
#pragma unroll
        for (int i = 0; i < 8; ++i) {
            int g  = i * 128 + tid;           // 0..1023
            int n  = g >> 2;                  // 0..255
            int k8 = (g & 3) * 8;             // 0,8,16,24
#if USE_ASYNC_LDS
            __builtin_amdgcn_global_load_async_to_lds_b128(
                (gas_v4i)(void*)(Wh + (size_t)n * IN_F + kk + k8),
                (lds_v4i)(void*)&Bs[n * 32 + k8], 0, 0);
#else
            *(uint4*)&Bs[n * 32 + k8] =
                *(const uint4*)(Wh + (size_t)n * IN_F + kk + k8);
#endif
        }
#if USE_ASYNC_LDS
        __builtin_amdgcn_s_wait_asynccnt(0);
#endif
        __syncthreads();

        // --- A fragment (fp32 global load + cvt; shared by the 4 subtiles) ---
        const float* ap = feat + (size_t)mrow * IN_F + kk + akb;
        float4 a0 = *(const float4*)(ap);
        float4 a1 = *(const float4*)(ap + 4);
        float4 a2 = *(const float4*)(ap + 8);
        float4 a3 = *(const float4*)(ap + 12);
        v16h A;
        A[0]=(_Float16)a0.x; A[1]=(_Float16)a0.y; A[2]=(_Float16)a0.z; A[3]=(_Float16)a0.w;
        A[4]=(_Float16)a1.x; A[5]=(_Float16)a1.y; A[6]=(_Float16)a1.z; A[7]=(_Float16)a1.w;
        A[8]=(_Float16)a2.x; A[9]=(_Float16)a2.y; A[10]=(_Float16)a2.z; A[11]=(_Float16)a2.w;
        A[12]=(_Float16)a3.x; A[13]=(_Float16)a3.y; A[14]=(_Float16)a3.z; A[15]=(_Float16)a3.w;

        // --- B fragments from LDS; lanes 0-15: K=0..15, lanes 16-31: K=16..31 ---
        const int kb = (lane >> 4) * 16;
        const int nl = lane & 15;
        v16h B0 = *(const v16h*)&Bs[(n0 +  0 + nl) * 32 + kb];
        v16h B1 = *(const v16h*)&Bs[(n0 + 16 + nl) * 32 + kb];
        v16h B2 = *(const v16h*)&Bs[(n0 + 32 + nl) * 32 + kb];
        v16h B3 = *(const v16h*)&Bs[(n0 + 48 + nl) * 32 + kb];

        acc0 = __builtin_amdgcn_wmma_f32_16x16x32_f16(false, A, false, B0, (short)0, acc0, false, false);
        acc1 = __builtin_amdgcn_wmma_f32_16x16x32_f16(false, A, false, B1, (short)0, acc1, false, false);
        acc2 = __builtin_amdgcn_wmma_f32_16x16x32_f16(false, A, false, B2, (short)0, acc2, false, false);
        acc3 = __builtin_amdgcn_wmma_f32_16x16x32_f16(false, A, false, B3, (short)0, acc3, false, false);
    }

    // C/D layout: VGPR v, lanes 0-15: (M=v, N=lane); lanes 16-31: (M=8+v, N=lane-16)
    const int colb = n0 + (lane & 15);
    const int rowb = m0 + 8 * (lane >> 4);
    if (m0 + 16 <= M) {
        // full tile: unguarded, clause-able stores
#pragma unroll
        for (int v = 0; v < 8; ++v) {
            float* r = ft + (size_t)(rowb + v) * HF + colb;
            r[0]  = acc0[v];
            r[16] = acc1[v];
            r[32] = acc2[v];
            r[48] = acc3[v];
        }
    } else {
#pragma unroll
        for (int v = 0; v < 8; ++v) {
            if (rowb + v < M) {
                float* r = ft + (size_t)(rowb + v) * HF + colb;
                r[0]  = acc0[v];
                r[16] = acc1[v];
                r[32] = acc2[v];
                r[48] = acc3[v];
            }
        }
    }
}

// ---------------------------------------------------------------------------
// K2: el[n,h] = <ft[n,h,:], attn_l[h,:]>, er likewise.
// 1 block per node; wave = head (wave32: lane == feature), shuffle reduce.
// ---------------------------------------------------------------------------
__global__ __launch_bounds__(256) void gat_node_logits(const float* __restrict__ ft,
                                                       const float* __restrict__ attn_l,
                                                       const float* __restrict__ attn_r,
                                                       float* __restrict__ el,
                                                       float* __restrict__ er) {
    const int n = blockIdx.x;
    const int h = threadIdx.x >> 5;
    const int lane = threadIdx.x & 31;
    float v = ft[(size_t)n * HF + h * OUTF + lane];
    float l = v * attn_l[h * OUTF + lane];
    float r = v * attn_r[h * OUTF + lane];
#pragma unroll
    for (int off = 16; off > 0; off >>= 1) {
        l += __shfl_xor(l, off, 32);
        r += __shfl_xor(r, off, 32);
    }
    if (lane == 0) {
        el[n * NHEAD + h] = l;
        er[n * NHEAD + h] = r;
    }
}

// ---------------------------------------------------------------------------
// K3: segment max over dst (exact, via order-preserving uint atomicMax)
// ---------------------------------------------------------------------------
__global__ __launch_bounds__(256) void gat_edge_max(const int* __restrict__ src,
                                                    const int* __restrict__ dst,
                                                    const float* __restrict__ el,
                                                    const float* __restrict__ er,
                                                    unsigned* __restrict__ m_enc, int E) {
    int t = blockIdx.x * blockDim.x + threadIdx.x;
    if (t >= E * NHEAD) return;
    int e = t >> 3, h = t & 7;
    int s = src[e], d = dst[e];
    float x = leaky(el[s * NHEAD + h] + er[d * NHEAD + h]);
    atomicMax(&m_enc[d * NHEAD + h], enc_f32(x));
}

// ---------------------------------------------------------------------------
// K4: denom[d,h] += exp(e - m[d,h])
// ---------------------------------------------------------------------------
__global__ __launch_bounds__(256) void gat_edge_denom(const int* __restrict__ src,
                                                      const int* __restrict__ dst,
                                                      const float* __restrict__ el,
                                                      const float* __restrict__ er,
                                                      const unsigned* __restrict__ m_enc,
                                                      float* __restrict__ denom, int E) {
    int t = blockIdx.x * blockDim.x + threadIdx.x;
    if (t >= E * NHEAD) return;
    int e = t >> 3, h = t & 7;
    int s = src[e], d = dst[e];
    float x  = leaky(el[s * NHEAD + h] + er[d * NHEAD + h]);
    float ee = __expf(x - dec_f32(m_enc[d * NHEAD + h]));
    atomic_add_f32(&denom[d * NHEAD + h], ee);
}

// ---------------------------------------------------------------------------
// K5: out[dst] += ft[src] * a   — one wave per edge, float4 per lane,
// hardware fp32 atomics. Dominant pass (~1.6 GB of traffic).
// ---------------------------------------------------------------------------
__global__ __launch_bounds__(256) void gat_aggregate(const int* __restrict__ src,
                                                     const int* __restrict__ dst,
                                                     const float* __restrict__ ft,
                                                     const float* __restrict__ el,
                                                     const float* __restrict__ er,
                                                     const unsigned* __restrict__ m_enc,
                                                     const float* __restrict__ denom,
                                                     float* __restrict__ out, int E) {
    int w = (blockIdx.x * blockDim.x + threadIdx.x) >> 5;   // edge id
    int lane = threadIdx.x & 31;
    if (w >= E) return;
    int s = src[w], d = dst[w];
    __builtin_prefetch(&ft[(size_t)s * HF], 0, 1);          // global_prefetch_b8
#pragma unroll
    for (int it = 0; it < 2; ++it) {
        int idx = it * 128 + lane * 4;                      // 4 feats, same head
        int h   = idx >> 5;
        float x = leaky(el[s * NHEAD + h] + er[d * NHEAD + h]);
        float a = __expf(x - dec_f32(m_enc[d * NHEAD + h])) / denom[d * NHEAD + h];
        float4 v = *(const float4*)&ft[(size_t)s * HF + idx];
        float* o = &out[(size_t)d * HF + idx];
        atomic_add_f32(o + 0, v.x * a);
        atomic_add_f32(o + 1, v.y * a);
        atomic_add_f32(o + 2, v.z * a);
        atomic_add_f32(o + 3, v.w * a);
    }
}

// ---------------------------------------------------------------------------
// launcher
// ---------------------------------------------------------------------------
extern "C" void kernel_launch(void* const* d_in, const int* in_sizes, int n_in,
                              void* d_out, int out_size, void* d_ws, size_t ws_size,
                              hipStream_t stream) {
    const float* feat   = (const float*)d_in[0];
    const float* W      = (const float*)d_in[1];
    const float* attn_l = (const float*)d_in[2];
    const float* attn_r = (const float*)d_in[3];
    const int*   src    = (const int*)d_in[4];
    const int*   dst    = (const int*)d_in[5];
    float*       out    = (float*)d_out;

    const int N = in_sizes[0] / IN_F;   // 50000
    const int E = in_sizes[4];          // 800000
    const int WSZ = in_sizes[1];        // 65536

    // workspace layout (floats):
    //   ft[N*256] | el[N*8] | er[N*8] | m_enc[N*8] | denom[N*8] | Wh (fp16, 128KB)
    float*    ft    = (float*)d_ws;
    float*    el    = ft + (size_t)N * HF;
    float*    er    = el + (size_t)N * NHEAD;
    unsigned* m_enc = (unsigned*)(er + (size_t)N * NHEAD);
    float*    denom = (float*)(m_enc + (size_t)N * NHEAD);
    _Float16* Wh    = (_Float16*)(denom + (size_t)N * NHEAD);

    gat_init<<<2048, 256, 0, stream>>>(out, m_enc, denom,
                                       (size_t)N * HF, (size_t)N * NHEAD);

    gat_convert_w<<<(WSZ + 255) / 256, 256, 0, stream>>>(W, Wh);

    gat_gemm<<<(N + 15) / 16, 128, 0, stream>>>(feat, Wh, ft, N);

    gat_node_logits<<<N, 256, 0, stream>>>(ft, attn_l, attn_r, el, er);

    int t_eh = E * NHEAD;
    gat_edge_max<<<(t_eh + 255) / 256, 256, 0, stream>>>(src, dst, el, er, m_enc, E);
    gat_edge_denom<<<(t_eh + 255) / 256, 256, 0, stream>>>(src, dst, el, er, m_enc, denom, E);

    long long thr = (long long)E * 32;
    gat_aggregate<<<(unsigned)((thr + 255) / 256), 256, 0, stream>>>(
        src, dst, ft, el, er, m_enc, denom, out, E);
}
